// TrajectoryScorer_67963562492452
// MI455X (gfx1250) — matrix operand order repaired and checked by reading, not compiled, not run on hardware
//
#include <hip/hip_runtime.h>
#include <hip/hip_bf16.h>

// ---------------- problem constants (from reference) ----------------
#define B_   32
#define K_   64
#define T_   80
#define C_   256
#define H_   200
#define W_   200
#define HW_  (H_ * W_)
#define BEV_RANGE_ 50.0f
#define LAM_PROG 1.0f
#define LAM_COLL 2.0f
#define LAM_COMF 0.5f

#define NTRAJ (B_ * K_)          // 2048 blocks
#define THREADS 128              // 4 wave32

typedef _Float16 v16h __attribute__((ext_vector_type(16)));
typedef float    v8f  __attribute__((ext_vector_type(8)));

__global__ __launch_bounds__(THREADS)
void TrajectoryScorer_kernel(const float* __restrict__ traj,   // (B,K,T,2)
                             const float* __restrict__ goals,  // (B,K,2)
                             const float* __restrict__ bev,    // (B,C,H,W)
                             const float* __restrict__ fe_w1,  // (2,64)
                             const float* __restrict__ fe_b1,  // (64)
                             const float* __restrict__ fe_w2,  // (64,128)
                             const float* __restrict__ fe_b2,  // (128)
                             const float* __restrict__ ch_w1,  // (384,64)
                             const float* __restrict__ ch_b1,  // (64)
                             const float* __restrict__ ch_w2,  // (64,1)
                             const float* __restrict__ ch_b2,  // (1)
                             float* __restrict__ out)          // 4*(B,K)
{
    const int bk   = blockIdx.x;          // trajectory id
    const int b    = bk >> 6;             // /K_
    const int tid  = threadIdx.x;
    const int lane = tid & 31;
    const int wv   = tid >> 5;            // wave id 0..3

    __shared__ float    trajS[T_ * 2];     // 640 B
    __shared__ int      idxS[T_];          // 320 B
    __shared__ _Float16 hS[T_ * 64];       // 10 KB : relu(x@w1+b1), f16 A-matrix
    __shared__ float    bevSum[C_];        // 1 KB
    __shared__ float    trajSum[128];      // 512 B
    __shared__ float    comb[384];         // 1.5 KB
    __shared__ float    red[64];
    __shared__ float    accSum;

    // ---- phase 0: init + stage trajectory ----
    if (tid < 128) trajSum[tid] = 0.0f;
    if (tid == 0)  accSum = 0.0f;
    for (int i = tid; i < T_ * 2; i += THREADS)
        trajS[i] = traj[(size_t)bk * (T_ * 2) + i];
    __syncthreads();

    // ---- phase 1: grid indices, comfort partials, first MLP layer -> f16 LDS ----
    if (tid < T_) {
        float px = trajS[tid * 2 + 0];
        float py = trajS[tid * 2 + 1];
        int gx = (int)((px / BEV_RANGE_ + 1.0f) * (W_ * 0.5f));  // trunc toward 0, like astype(int32)
        int gy = (int)((py / BEV_RANGE_ + 1.0f) * (H_ * 0.5f));
        gx = min(max(gx, 0), W_ - 1);
        gy = min(max(gy, 0), H_ - 1);
        idxS[tid] = gy * W_ + gx;
    }
    if (tid < T_ - 2) {
        int t = tid;
        float ax = trajS[(t + 2) * 2 + 0] - 2.0f * trajS[(t + 1) * 2 + 0] + trajS[t * 2 + 0];
        float ay = trajS[(t + 2) * 2 + 1] - 2.0f * trajS[(t + 1) * 2 + 1] + trajS[t * 2 + 1];
        atomicAdd(&accSum, sqrtf(ax * ax + ay * ay));   // ds_add_f32
    }
    // h[t][j] = relu(px*w1[0][j] + py*w1[1][j] + b1[j]);  80*64 = 5120 values
    for (int i = tid; i < T_ * 64; i += THREADS) {
        int t = i >> 6, j = i & 63;
        float v = trajS[t * 2 + 0] * fe_w1[j] + trajS[t * 2 + 1] * fe_w1[64 + j] + fe_b1[j];
        hS[i] = (_Float16)(v > 0.0f ? v : 0.0f);
    }
    __syncthreads();

    // ---- phase 2: WMMA  y = relu(h @ w2 + b2), column-summed over T ----
    // wave wv owns N-tiles {2*wv, 2*wv+1}; K = 64 = 2 chunks of 32; M = 80 = 5 tiles.
    {
        const int nloc  = lane & 15;
        const int khalf = (lane >> 4) & 1;        // lane group
        // ---- B fragments: 32x16 f16, lanes 0-15 hold K=0..15, lanes 16-31 K=16..31
        v16h bf[2][2];                            // [ntile][kchunk]
        #pragma unroll
        for (int j = 0; j < 2; ++j) {
            const int n = (wv * 2 + j) * 16 + nloc;
            #pragma unroll
            for (int kc = 0; kc < 2; ++kc) {
                const int kbase = kc * 32 + khalf * 16;
                #pragma unroll
                for (int s = 0; s < 16; ++s)
                    bf[j][kc][s] = (_Float16)fe_w2[(size_t)(kbase + s) * 128 + n];
            }
        }
        float bias[2];
        #pragma unroll
        for (int j = 0; j < 2; ++j) bias[j] = fe_b2[(wv * 2 + j) * 16 + nloc];

        for (int mt = 0; mt < 5; ++mt) {
            // ---- A fragments: 16x32 f16; lane<16: K=0..15, lane>=16: K=16..31
            const int m    = mt * 16 + nloc;
            const int koff = khalf * 8;
            v16h a0, a1;
            #pragma unroll
            for (int s = 0; s < 16; ++s) {
                const int k = s + koff * 2;
                a0[s] = hS[m * 64 + k];
                a1[s] = hS[m * 64 + 32 + k];
            }
            #pragma unroll
            for (int j = 0; j < 2; ++j) {
                v8f acc = {};
                acc = __builtin_amdgcn_wmma_f32_16x16x32_f16(false, a0, false, bf[j][0],
                                                             (short)0, acc, false, false);
                acc = __builtin_amdgcn_wmma_f32_16x16x32_f16(false, a1, false, bf[j][1],
                                                             (short)0, acc, false, false);
                // D layout: VGPR r -> M = mt*16 + r (+8 for lanes 16-31), N = tile_n + nloc.
                float partial = 0.0f;
                #pragma unroll
                for (int r = 0; r < 8; ++r) {
                    float v = acc[r] + bias[j];
                    partial += (v > 0.0f) ? v : 0.0f;
                }
                atomicAdd(&trajSum[(wv * 2 + j) * 16 + nloc], partial);
            }
        }
    }

    // ---- phase 3: BEV gather-average (each thread owns 2 channels) ----
    {
        const float* bevB = bev + (size_t)b * C_ * HW_;
        const int c0 = tid, c1 = tid + 128;
        float s0 = 0.0f, s1 = 0.0f;
        for (int t = 0; t < T_; ++t) {
            const int id = idxS[t];
            s0 += bevB[(size_t)c0 * HW_ + id];
            s1 += bevB[(size_t)c1 * HW_ + id];
        }
        bevSum[c0] = s0 * (1.0f / T_);
        bevSum[c1] = s1 * (1.0f / T_);
    }
    __syncthreads();

    // ---- phase 4: combined = [traj_avg(128), bev_avg(256)] ----
    for (int i = tid; i < 384; i += THREADS)
        comb[i] = (i < 128) ? trajSum[i] * (1.0f / T_) : bevSum[i - 128];
    __syncthreads();

    // ---- phase 5: classifier head (VALU dot products) ----
    if (tid < 64) {
        float acc = ch_b1[tid];
        for (int i = 0; i < 384; ++i)
            acc = fmaf(comb[i], ch_w1[(size_t)i * 64 + tid], acc);
        float hr = (acc > 0.0f) ? acc : 0.0f;
        red[tid] = hr * ch_w2[tid];
    }
    __syncthreads();

    // ---- phase 6: finalize (thread 0) ----
    if (tid == 0) {
        float logit = ch_b2[0];
        for (int j = 0; j < 64; ++j) logit += red[j];
        const float collision = 1.0f / (1.0f + __expf(-logit));

        const float fx = trajS[(T_ - 1) * 2 + 0];
        const float fy = trajS[(T_ - 1) * 2 + 1];
        const float gx = goals[(size_t)bk * 2 + 0];
        const float gy = goals[(size_t)bk * 2 + 1];
        const float dist = sqrtf((fx - gx) * (fx - gx) + (fy - gy) * (fy - gy));
        const float progress = 1.0f / (1.0f + dist);

        const float comfort = 1.0f / (1.0f + accSum * (1.0f / (T_ - 2)));

        const float score = LAM_PROG * progress - LAM_COLL * collision + LAM_COMF * comfort;
        out[bk]             = score;
        out[NTRAJ     + bk] = progress;
        out[2 * NTRAJ + bk] = collision;
        out[3 * NTRAJ + bk] = comfort;
    }
}

extern "C" void kernel_launch(void* const* d_in, const int* in_sizes, int n_in,
                              void* d_out, int out_size, void* d_ws, size_t ws_size,
                              hipStream_t stream) {
    (void)in_sizes; (void)n_in; (void)d_ws; (void)ws_size; (void)out_size;
    const float* traj  = (const float*)d_in[0];
    const float* goals = (const float*)d_in[1];
    const float* bev   = (const float*)d_in[2];
    const float* fe_w1 = (const float*)d_in[3];
    const float* fe_b1 = (const float*)d_in[4];
    const float* fe_w2 = (const float*)d_in[5];
    const float* fe_b2 = (const float*)d_in[6];
    const float* ch_w1 = (const float*)d_in[7];
    const float* ch_b1 = (const float*)d_in[8];
    const float* ch_w2 = (const float*)d_in[9];
    const float* ch_b2 = (const float*)d_in[10];
    float* out = (float*)d_out;

    hipLaunchKernelGGL(TrajectoryScorer_kernel,
                       dim3(NTRAJ), dim3(THREADS), 0, stream,
                       traj, goals, bev, fe_w1, fe_b1, fe_w2, fe_b2,
                       ch_w1, ch_b1, ch_w2, ch_b2, out);
}
